// UAVEnhancedModel_25194278159154
// MI455X (gfx1250) — matrix operand order repaired and checked by reading, not compiled, not run on hardware
//
#include <hip/hip_runtime.h>
#include <cstddef>
#include <cstdint>

typedef __attribute__((ext_vector_type(16))) _Float16 v16h;
typedef __attribute__((ext_vector_type(8)))  _Float16 v8h;
typedef __attribute__((ext_vector_type(8)))  float    v8f;
typedef __attribute__((ext_vector_type(4)))  unsigned int v4ui;
typedef __attribute__((ext_vector_type(8)))  int v8ii;
typedef __attribute__((ext_vector_type(4)))  int v4ii;

#define S_LEN 4096
#define I_DIM 512
#define H_DIM 1024
#define G_DIM 4096    // 4*H
#define NWG_SCAN 64   // persistent scan WGs (H/16)
#define LDS_ROW 1040  // padded W_hh row stride in halves (1024 + 2x8 pad)

__device__ __forceinline__ float sigmoidf_(float x) { return 1.0f / (1.0f + expf(-x)); }

// ---- WMMA tile loaders (layouts per CDNA5 ISA 7.12.2, 16-bit A / B / f32 C) ----
__device__ __forceinline__ v16h load_a_tile(const _Float16* __restrict__ row, int kb, int hs) {
  const v8h lo = *(const v8h*)(row + kb + hs * 8);
  const v8h hi = *(const v8h*)(row + kb + 16 + hs * 8);
  v16h a;
#pragma unroll
  for (int e = 0; e < 8; ++e) { a[e] = lo[e]; a[e + 8] = hi[e]; }
  return a;
}
__device__ __forceinline__ v16h load_b_tile(const _Float16* __restrict__ colrow, int kb, int hs) {
  const v8h lo = *(const v8h*)(colrow + kb + hs * 8);
  const v8h hi = *(const v8h*)(colrow + kb + 16 + hs * 8);
  v16h b;
#pragma unroll
  for (int e = 0; e < 8; ++e) { b[e] = lo[e]; b[e + 8] = hi[e]; }
  return b;
}

// ---- C[MxN] = A[MxK] @ W[NxK]^T + bias[N]; one wave per 16x64 tile, K static,
// ---- fully unrolled + double-buffered so loads overlap the 4 WMMAs ----
template <int K>
__global__ void gemm_wmma_kernel(const _Float16* __restrict__ A,
                                 const _Float16* __restrict__ W,
                                 const float* __restrict__ bias,
                                 float* __restrict__ C,
                                 int M, int N) {
  const int wavesPerBlk = blockDim.x >> 5;
  const int waveId = blockIdx.x * wavesPerBlk + ((int)threadIdx.x >> 5);
  const int ng = N >> 6;                     // groups of 4 N-tiles
  const int mt = waveId / ng;
  const int ntg = waveId - mt * ng;
  if (mt >= (M >> 4)) return;                // wave-uniform: EXEC all-1 at WMMA
  const int lane = threadIdx.x & 31;
  const int r = lane & 15, hs = lane >> 4;
  const _Float16* arow = A + (size_t)(mt * 16 + r) * K;
  const _Float16* w0 = W + (size_t)(ntg * 64 + r) * K;
  const _Float16* w1 = w0 + (size_t)16 * K;
  const _Float16* w2 = w0 + (size_t)32 * K;
  const _Float16* w3 = w0 + (size_t)48 * K;
  v8f a0 = {}, a1 = {}, a2 = {}, a3 = {};

  // prologue: load k-chunk 0
  v16h ca  = load_a_tile(arow, 0, hs);
  v16h cb0 = load_b_tile(w0, 0, hs);
  v16h cb1 = load_b_tile(w1, 0, hs);
  v16h cb2 = load_b_tile(w2, 0, hs);
  v16h cb3 = load_b_tile(w3, 0, hs);
#pragma unroll
  for (int kb = 0; kb < K; kb += 32) {
    v16h na, nb0, nb1, nb2, nb3;
    if (kb + 32 < K) {                       // issue next chunk's loads first
      na  = load_a_tile(arow, kb + 32, hs);
      nb0 = load_b_tile(w0, kb + 32, hs);
      nb1 = load_b_tile(w1, kb + 32, hs);
      nb2 = load_b_tile(w2, kb + 32, hs);
      nb3 = load_b_tile(w3, kb + 32, hs);
    }
    a0 = __builtin_amdgcn_wmma_f32_16x16x32_f16(false, ca, false, cb0, (short)0, a0, false, false);
    a1 = __builtin_amdgcn_wmma_f32_16x16x32_f16(false, ca, false, cb1, (short)0, a1, false, false);
    a2 = __builtin_amdgcn_wmma_f32_16x16x32_f16(false, ca, false, cb2, (short)0, a2, false, false);
    a3 = __builtin_amdgcn_wmma_f32_16x16x32_f16(false, ca, false, cb3, (short)0, a3, false, false);
    if (kb + 32 < K) { ca = na; cb0 = nb0; cb1 = nb1; cb2 = nb2; cb3 = nb3; }
  }
  const int nb = ntg * 64 + r;
  const float bv0 = bias ? bias[nb]      : 0.0f;
  const float bv1 = bias ? bias[nb + 16] : 0.0f;
  const float bv2 = bias ? bias[nb + 32] : 0.0f;
  const float bv3 = bias ? bias[nb + 48] : 0.0f;
#pragma unroll
  for (int v = 0; v < 8; ++v) {
    float* crow = C + (size_t)(mt * 16 + hs * 8 + v) * N + nb;   // C: M = hs*8+v, N = lane&15
    crow[0]  = a0[v] + bv0;
    crow[16] = a1[v] + bv1;
    crow[32] = a2[v] + bv2;
    crow[48] = a3[v] + bv3;
  }
}

// ---------- small elementwise / setup kernels ----------
__global__ void cvt_f16_kernel(const float* __restrict__ src, _Float16* __restrict__ dst, int n) {
  int i = blockIdx.x * blockDim.x + threadIdx.x;
  if (i < n) dst[i] = (_Float16)src[i];
}

__global__ void init_kernel(const float* __restrict__ b_ih, const float* __restrict__ b_hh,
                            float* __restrict__ bcomb, _Float16* __restrict__ h16,
                            float* __restrict__ cbuf, unsigned int* __restrict__ barctr) {
  int i = blockIdx.x * blockDim.x + threadIdx.x;
  if (i < G_DIM) bcomb[i] = b_ih[i] + b_hh[i];
  if (i < H_DIM) { h16[i] = (_Float16)0.0f; cbuf[i] = 0.0f; }
  if (i == 0) *barctr = 0u;
}

__global__ void decay_kernel(const float* __restrict__ deltas, const float* __restrict__ Wd,
                             float* __restrict__ dec) {
  int i = blockIdx.x * blockDim.x + threadIdx.x;
  if (i < S_LEN * H_DIM) {
    int s = i >> 10, h = i & (H_DIM - 1);
    float w = fminf(fmaxf(Wd[h], 1e-7f), 10.0f);
    dec[i] = expf(-deltas[s] * w);
  }
}

__global__ void u1_kernel(const float* __restrict__ uavf, const float* __restrict__ W1,
                          const float* __restrict__ b1, _Float16* __restrict__ u1) {
  int i = blockIdx.x * blockDim.x + threadIdx.x;
  if (i < S_LEN * H_DIM) {
    int s = i >> 10, h = i & (H_DIM - 1);
    float v = uavf[s] * W1[h] + b1[h];
    u1[i] = (_Float16)fmaxf(v, 0.0f);
  }
}

// ---------- persistent LSTM scan ----------
__device__ __forceinline__ void grid_barrier(unsigned int* __restrict__ ctr, unsigned int round) {
  __threadfence();                 // release this WG's stores
  __syncthreads();
  if (threadIdx.x == 0) {
    atomicAdd(ctr, 1u);
    const unsigned int target = round * (unsigned int)NWG_SCAN;   // monotonic: no reset race
    while (atomicAdd(ctr, 0u) < target) { __builtin_amdgcn_s_sleep(2); }
  }
  __syncthreads();
  __threadfence();                 // acquire: drop stale WGP-cache lines
}

// Padded-LDS offset: rows stride LDS_ROW halves; +8 halves after first 512 (16B pad per 256 DW)
__device__ __forceinline__ int lds_koff(int kb) { return kb + ((kb >= 512) ? 8 : 0); }

__global__ void __launch_bounds__(128, 1)
scan_kernel(const _Float16* __restrict__ Whh16,   // [4096 x 1024] f16 row-major
            const float* __restrict__ xproj,      // [S x 4096] (bias folded in)
            const float* __restrict__ decays,     // [S x 1024]
            const float* __restrict__ uf,         // [S x 1024]
            const float* __restrict__ ln_g,
            const float* __restrict__ ln_b,
            _Float16* __restrict__ h16,           // [1024] recurrent h (f16)
            float* __restrict__ cbuf,             // [1024] cell state
            float* __restrict__ hln,              // [S x 1024] LN outputs
            float* __restrict__ redpart,          // [NWG_SCAN*2] reduction slots
            unsigned int* __restrict__ barctr) {
  const int wg = blockIdx.x;              // owns h-indices [wg*16, wg*16+16)
  const int tid = threadIdx.x;            // 128 thr = 4 waves; wave = gate block
  const int gate = tid >> 5;
  const int lane = tid & 31;
  const int r = lane & 15, hs = lane >> 4;
  const int jbase = wg * 16;

  extern __shared__ _Float16 smem[];
  _Float16* whh_lds = smem;                       // 64 padded rows x LDS_ROW
  _Float16* h_lds   = smem + 64 * LDS_ROW;        // 1024

  // ---- one-time: stage this WG's 64x1024 W_hh slice into LDS via TDM ----
#if __has_builtin(__builtin_amdgcn_tensor_load_to_lds)
  if (tid == 0) {
    const uint64_t ga = (uint64_t)(uintptr_t)(Whh16 + (size_t)jbase * H_DIM);
    const unsigned lds_base = (unsigned)(uintptr_t)whh_lds;
    // D# group0: count=1, lds_addr, 57b global_addr, type=2 (bits 127:126)
    v4ui g0 = { 1u, lds_base, (unsigned)(ga & 0xFFFFFFFFu),
                (unsigned)(((ga >> 32) & 0x01FFFFFFu) | 0x80000000u) };
    // group1: data_size=2B, pad_enable, pad_interval=256DW, pad_amount=4DW;
    // tensor_dim0=1024, tensor_dim1=16, tile=1024x16x4, dim0_stride=1024, dim1_stride=2^20
    v8ii g1 = { (int)0x07D10000,
                (int)(1024u << 16),        // tensor_dim0[15:0] @ bits63:48
                (int)(16u << 16),          // tensor_dim1[15:0] @ bits95:80
                (int)(1024u << 16),        // tile_dim0 @ bits127:112
                (int)((4u << 16) | 16u),   // tile_dim1=16, tile_dim2=4
                (int)1024,                 // tensor_dim0_stride lo32
                0,                         // stride0 hi16 / stride1 lo16
                (int)0x10 };               // tensor_dim1_stride[47:16] -> 0x100000
    v4ii g2 = { 4, 0, 0, 0 };              // tensor_dim2=4
    v4ii g3 = { 0, 0, 0, 0 };
#if defined(__clang_major__) && (__clang_major__ >= 23)
    v8ii g4 = { 0, 0, 0, 0, 0, 0, 0, 0 };
    __builtin_amdgcn_tensor_load_to_lds(g0, g1, g2, g3, g4, 0);
#else
    __builtin_amdgcn_tensor_load_to_lds(g0, g1, g2, g3, 0);
#endif
    __builtin_amdgcn_s_wait_tensorcnt(0);
  }
#else
  for (int it = tid; it < 64 * 128; it += 128) {   // manual fallback, same padded layout
    const int row = it >> 7, ch = it & 127;
    const int gblk = row >> 4, y = row & 15;
    const int k = ch * 8;
    const _Float16* src = Whh16 + (size_t)(gblk * H_DIM + jbase + y) * H_DIM + k;
    *(v8h*)(whh_lds + (size_t)row * LDS_ROW + lds_koff(k & ~31) + (k & 31)) = *(const v8h*)src;
  }
#endif
  __syncthreads();

  __shared__ float lds_gates[4][16];
  __shared__ float lds_hraw[16];

  const _Float16* wrow_lds = whh_lds + (size_t)(gate * 16 + r) * LDS_ROW;

  unsigned int round = 0;
  for (int t = 0; t < S_LEN; ++t) {
    // stage h into LDS (128 x 16B = 2KB), broadcast B-operand source
    *(v8h*)(h_lds + tid * 8) = *(const v8h*)(h16 + tid * 8);
    __syncthreads();

    // ---- phase 1: gates = W_hh @ h (A from LDS, B = h broadcast into 16 cols) ----
    v8f acc = {};
#pragma unroll
    for (int kb = 0; kb < H_DIM; kb += 32) {
      v16h a = load_a_tile(wrow_lds, lds_koff(kb), hs);
      v16h b = load_b_tile(h_lds, kb, hs);
      acc = __builtin_amdgcn_wmma_f32_16x16x32_f16(false, a, false, b, (short)0, acc, false, false);
    }
    if (r == 0) {                           // lanes 0 / 16: column n==0, rows hs*8+v
      const float* xp = xproj + (size_t)t * G_DIM + gate * H_DIM + jbase + hs * 8;
#pragma unroll
      for (int v = 0; v < 8; ++v) lds_gates[gate][hs * 8 + v] = acc[v] + xp[v];
    }
    __syncthreads();

    // ---- LSTM cell math on 16 h-lanes; deterministic partial LN reduction ----
    if (tid < 16) {
      const int jl = tid, jg = jbase + jl;
      const float gi = lds_gates[0][jl], gf = lds_gates[1][jl];
      const float gg = lds_gates[2][jl], go = lds_gates[3][jl];
      const float chat = cbuf[jg] * decays[(size_t)t * H_DIM + jg];
      float cn = sigmoidf_(gf) * chat + sigmoidf_(gi) * tanhf(gg);
      float hn = sigmoidf_(go) * tanhf(cn) + 1.5f * uf[(size_t)t * H_DIM + jg];
      hn = fminf(fmaxf(hn, -10.0f), 10.0f);
      cn = fminf(fmaxf(cn, -10.0f), 10.0f);
      cbuf[jg] = cn;
      lds_hraw[jl] = hn;
      float s1 = hn, s2 = hn * hn;
#pragma unroll
      for (int off = 8; off >= 1; off >>= 1) {
        s1 += __shfl_down(s1, off, 32);
        s2 += __shfl_down(s2, off, 32);
      }
      if (jl == 0) { redpart[2 * wg] = s1; redpart[2 * wg + 1] = s2; }
    }
    ++round;
    grid_barrier(barctr, round);            // A: all partials published

    // ---- phase 2: LayerNorm with fixed-order (deterministic) global sum ----
    if (tid < 16) {
      float S1 = 0.0f, S2 = 0.0f;
      for (int q = 0; q < NWG_SCAN; ++q) { S1 += redpart[2 * q]; S2 += redpart[2 * q + 1]; }
      const float mu = S1 * (1.0f / (float)H_DIM);
      const float var = S2 * (1.0f / (float)H_DIM) - mu * mu;
      const float inv = rsqrtf(var + 1e-5f);
      const int jl = tid, jg = jbase + jl;
      const float hl = (lds_hraw[jl] - mu) * inv * ln_g[jg] + ln_b[jg];
      hln[(size_t)t * H_DIM + jg] = hl;
      h16[jg] = (_Float16)hl;
    }
    ++round;
    grid_barrier(barctr, round);            // B: h fully updated before next step
  }
}

// ---------- head + tail ----------
__global__ void head_kernel(const float* __restrict__ hln, const float* __restrict__ ufcW,
                            const float* __restrict__ ufcb, float* __restrict__ out) {
  const int s = blockIdx.x * ((int)blockDim.x >> 5) + ((int)threadIdx.x >> 5);
  const int lane = threadIdx.x & 31;
  if (s >= S_LEN) return;
  float acc = 0.0f;
  for (int k = lane; k < H_DIM; k += 32) acc += hln[(size_t)s * H_DIM + k] * ufcW[k];
#pragma unroll
  for (int off = 16; off >= 1; off >>= 1) acc += __shfl_down(acc, off, 32);
  if (lane == 0) out[s] = sigmoidf_(acc + ufcb[0]);
}

__global__ void tail_kernel(const float* __restrict__ hln, const float* __restrict__ cbuf,
                            float* __restrict__ out) {
  int j = blockIdx.x * blockDim.x + threadIdx.x;
  if (j < H_DIM) {
    out[S_LEN + j] = hln[(size_t)(S_LEN - 1) * H_DIM + j];   // h_last
    out[S_LEN + H_DIM + j] = cbuf[j];                        // c_last
  }
}

// ---------------- workspace layout (256B-aligned offsets) ----------------
static constexpr size_t OFF_WHH16 = 0;
static constexpr size_t OFF_X16   = OFF_WHH16 + (size_t)G_DIM * H_DIM * 2;
static constexpr size_t OFF_WIH16 = OFF_X16 + (size_t)S_LEN * I_DIM * 2;
static constexpr size_t OFF_W216  = OFF_WIH16 + (size_t)G_DIM * I_DIM * 2;
static constexpr size_t OFF_U116  = OFF_W216 + (size_t)H_DIM * H_DIM * 2;
static constexpr size_t OFF_H16   = OFF_U116 + (size_t)S_LEN * H_DIM * 2;
static constexpr size_t OFF_BCOMB = OFF_H16 + 4096;
static constexpr size_t OFF_DEC   = OFF_BCOMB + (size_t)G_DIM * 4;
static constexpr size_t OFF_UF    = OFF_DEC + (size_t)S_LEN * H_DIM * 4;
static constexpr size_t OFF_XPROJ = OFF_UF + (size_t)S_LEN * H_DIM * 4;
static constexpr size_t OFF_HLN   = OFF_XPROJ + (size_t)S_LEN * G_DIM * 4;
static constexpr size_t OFF_CBUF  = OFF_HLN + (size_t)S_LEN * H_DIM * 4;
static constexpr size_t OFF_RED   = OFF_CBUF + 4096;
static constexpr size_t OFF_BAR   = OFF_RED + 1024;

extern "C" void kernel_launch(void* const* d_in, const int* in_sizes, int n_in,
                              void* d_out, int out_size, void* d_ws, size_t ws_size,
                              hipStream_t stream) {
  const float* x      = (const float*)d_in[0];
  const float* deltas = (const float*)d_in[1];
  const float* uavf   = (const float*)d_in[2];
  const float* W_ih   = (const float*)d_in[3];
  const float* W_hh   = (const float*)d_in[4];
  const float* b_ih   = (const float*)d_in[5];
  const float* b_hh   = (const float*)d_in[6];
  const float* Wd     = (const float*)d_in[7];
  const float* uavW1  = (const float*)d_in[8];
  const float* uavb1  = (const float*)d_in[9];
  const float* uavW2  = (const float*)d_in[10];
  const float* uavb2  = (const float*)d_in[11];
  const float* ufcW   = (const float*)d_in[12];
  const float* ufcb   = (const float*)d_in[13];
  const float* ln_g   = (const float*)d_in[14];
  const float* ln_b   = (const float*)d_in[15];
  (void)in_sizes; (void)n_in; (void)out_size; (void)ws_size;

  char* ws = (char*)d_ws;
  _Float16* Whh16  = (_Float16*)(ws + OFF_WHH16);
  _Float16* x16    = (_Float16*)(ws + OFF_X16);
  _Float16* Wih16  = (_Float16*)(ws + OFF_WIH16);
  _Float16* W216   = (_Float16*)(ws + OFF_W216);
  _Float16* u116   = (_Float16*)(ws + OFF_U116);
  _Float16* h16    = (_Float16*)(ws + OFF_H16);
  float*    bcomb  = (float*)(ws + OFF_BCOMB);
  float*    dec    = (float*)(ws + OFF_DEC);
  float*    uf     = (float*)(ws + OFF_UF);
  float*    xproj  = (float*)(ws + OFF_XPROJ);
  float*    hln    = (float*)(ws + OFF_HLN);
  float*    cbuf   = (float*)(ws + OFF_CBUF);
  float*    red    = (float*)(ws + OFF_RED);
  unsigned int* bar = (unsigned int*)(ws + OFF_BAR);
  float* out = (float*)d_out;

  const int TB = 256;
  cvt_f16_kernel<<<(G_DIM * H_DIM + TB - 1) / TB, TB, 0, stream>>>(W_hh, Whh16, G_DIM * H_DIM);
  cvt_f16_kernel<<<(S_LEN * I_DIM + TB - 1) / TB, TB, 0, stream>>>(x, x16, S_LEN * I_DIM);
  cvt_f16_kernel<<<(G_DIM * I_DIM + TB - 1) / TB, TB, 0, stream>>>(W_ih, Wih16, G_DIM * I_DIM);
  cvt_f16_kernel<<<(H_DIM * H_DIM + TB - 1) / TB, TB, 0, stream>>>(uavW2, W216, H_DIM * H_DIM);
  init_kernel<<<(G_DIM + TB - 1) / TB, TB, 0, stream>>>(b_ih, b_hh, bcomb, h16, cbuf, bar);
  decay_kernel<<<(S_LEN * H_DIM + TB - 1) / TB, TB, 0, stream>>>(deltas, Wd, dec);
  u1_kernel<<<(S_LEN * H_DIM + TB - 1) / TB, TB, 0, stream>>>(uavf, uavW1, uavb1, u116);

  // GEMM1: x_proj[S x 4H] = x @ W_ih^T + (b_ih+b_hh); 16x64 per wave, 8 waves/block
  gemm_wmma_kernel<I_DIM><<<(S_LEN / 16) * (G_DIM / 64) / 8, 256, 0, stream>>>(
      x16, Wih16, bcomb, xproj, S_LEN, G_DIM);
  // GEMM2: uf[S x H] = relu_stage @ uavW2^T + uavb2
  gemm_wmma_kernel<H_DIM><<<(S_LEN / 16) * (H_DIM / 64) / 8, 256, 0, stream>>>(
      u116, W216, uavb2, uf, S_LEN, H_DIM);

  // persistent recurrent scan: dynamic LDS = 64*LDS_ROW + 1024 halves
  const size_t scan_lds = (size_t)(64 * LDS_ROW + 1024) * sizeof(_Float16);
  scan_kernel<<<NWG_SCAN, 128, scan_lds, stream>>>(Whh16, xproj, dec, uf, ln_g, ln_b,
                                                   h16, cbuf, hln, red, bar);

  head_kernel<<<(S_LEN / 8), 256, 0, stream>>>(hln, ufcW, ufcb, out);
  tail_kernel<<<(H_DIM + TB - 1) / TB, TB, 0, stream>>>(hln, cbuf, out);
}
